// ItemTower_36223754175138
// MI455X (gfx1250) — matrix operand order repaired and checked by reading, not compiled, not run on hardware
//
#include <hip/hip_runtime.h>
#include <math.h>

// ---------------------------------------------------------------------------
// Types
// ---------------------------------------------------------------------------
typedef _Float16 h16;
typedef __attribute__((ext_vector_type(16))) _Float16 v16h;
typedef __attribute__((ext_vector_type(8)))  _Float16 v8h;
typedef __attribute__((ext_vector_type(8)))  float    v8f;

#define DEVINL __device__ __forceinline__

// ---------------------------------------------------------------------------
// Packed f16 weight arena layout (element offsets, units = halfs).
// Each matrix stored as [Kp/32][Np/16] tiles of 512 halfs (32 lanes x 16 halfs)
// in the v_wmma 16-bit B-operand lane layout.
// ---------------------------------------------------------------------------
constexpr size_t OFF_WT1  = 0;                          // 384 x 192
constexpr size_t OFF_WT2  = OFF_WT1  + 384*192;         // 192 x 96
constexpr size_t OFF_WB0  = OFF_WT2  + 192*96;          // 224 x 384 (K 208 pad)
constexpr size_t OFF_WB1  = OFF_WB0  + 224*384;         // 384 x 256
constexpr size_t OFF_WB2  = OFF_WB1  + 384*256;         // 256 x 128
constexpr size_t OFF_WA   = OFF_WB2  + 256*128;         // 128 x 32  (N 18 pad)
constexpr size_t OFF_E1   = OFF_WA   + 128*32;          // 18 x (128 x 64)
constexpr size_t OFF_E2   = OFF_E1   + 18*128*64;       // 18 x (64 x 32)
constexpr size_t OFF_E3   = OFF_E2   + 18*64*32;        // 18 x (32 x 32)
constexpr size_t OFF_WAGG = OFF_E3   + 18*32*32;        // 160 x 128
constexpr size_t OFF_WO   = OFF_WAGG + 160*128;         // 128 x 128
constexpr size_t PACK_TOTAL = OFF_WO + 128*128;         // 552960 halfs (~1.1 MB)

// ---------------------------------------------------------------------------
// Per-wave LDS arena (M = 32 rows per wave). Padded row strides so the 16
// rows of an A-fragment ds_load_b128 hit distinct bank groups.
// ---------------------------------------------------------------------------
constexpr int MROWS    = 32;   // rows per wave (two 16-row WMMA sub-tiles)
constexpr int BSTRIDE  = 392;  // for 384-wide activation buffers (784B/row)
constexpr int H1STRIDE = 72;   // for 64-wide expert hidden
constexpr int H2STRIDE = 40;   // for 32-wide expert hidden

struct alignas(16) WaveLds {
  h16   buf0[MROWS * BSTRIDE];   // 25088 B
  h16   buf1[MROWS * BSTRIDE];   // 25088 B
  h16   h1[MROWS * H1STRIDE];    // 4608 B (also reused as float logits[32][32])
  h16   h2[MROWS * H2STRIDE];    // 2560 B
  float wls[MROWS * 18];         // 2304 B : attn * gv
};                                // total 59648 B < 64 KB

// ---------------------------------------------------------------------------
// Fragment helpers
// ---------------------------------------------------------------------------
DEVINL v8f vzero8() {
  v8f z;
#pragma unroll
  for (int i = 0; i < 8; ++i) z[i] = 0.f;
  return z;
}

DEVINL v8f wmma_f16(v16h a, v16h b, v8f c) {
  return __builtin_amdgcn_wmma_f32_16x16x32_f16(
      /*neg_a=*/false, a, /*neg_b=*/false, b,
      /*c_mod=*/(short)0, c, /*reuse_a=*/false, /*reuse_b=*/false);
}

// A fragment (16x32 f16, M x K) from row-major LDS [16][stride]:
// lane l: M = l&15; e<8 -> K = k0 + hi8 + e ; e>=8 -> K = k0 + 16 + hi8 + (e-8)
DEVINL v16h load_a(const h16* base, int stride, int k0, int lane) {
  const int row = lane & 15;
  const int hi8 = (lane & 16) ? 8 : 0;
  const h16* p = base + row * stride + k0 + hi8;
  union { v16h v; v8h h[2]; } u;
  u.h[0] = *(const v8h*)p;          // ds_load_b128
  u.h[1] = *(const v8h*)(p + 16);   // ds_load_b128
  return u.v;
}

// B fragment (32x16 f16, K x N) from packed global arena: one contiguous 32B
// chunk per lane.
DEVINL v16h load_b(const h16* packed, int nbt, int kb, int nb, int lane) {
  return *(const v16h*)(packed + ((size_t)(kb * nbt + nb) << 9) + (lane << 4));
}

// One full layer over 32 rows: two 16-row sub-tiles share every B fragment,
// halving L2 weight traffic per row.
// out[0:32, ocol0 : ocol0+Nout] = act(A[0:32, :Kp] @ B + bias)
DEVINL void gemm_layer(const h16* A, int astride, int Kp,
                       const h16* Bp, int Nout,
                       h16* outb, int ostride, int ocol0,
                       const float* __restrict__ bias, bool relu, int lane) {
  const int nbt = Nout >> 4;
  const int kbt = Kp >> 5;
  const int cl = lane & 15;
  const int rb = (lane & 16) ? 8 : 0;
  const h16* A1 = A + 16 * astride;
  for (int nb = 0; nb < nbt; ++nb) {
    v8f acc0 = vzero8();
    v8f acc1 = vzero8();
    for (int kb = 0; kb < kbt; ++kb) {
      v16h a0 = load_a(A,  astride, kb << 5, lane);
      v16h a1 = load_a(A1, astride, kb << 5, lane);
      v16h b  = load_b(Bp, nbt, kb, nb, lane);
      acc0 = wmma_f16(a0, b, acc0);
      acc1 = wmma_f16(a1, b, acc1);
    }
    const float bv = bias[nb * 16 + cl];
    h16* op = outb + ocol0 + nb * 16 + cl;
#pragma unroll
    for (int r = 0; r < 8; ++r) {
      float v0 = acc0[r] + bv;
      float v1 = acc1[r] + bv;
      if (relu) { v0 = fmaxf(v0, 0.f); v1 = fmaxf(v1, 0.f); }
      op[(rb + r) * ostride]        = (h16)v0;
      op[(16 + rb + r) * ostride]   = (h16)v1;
    }
  }
}

// Row-wise LayerNorm over Nw cols: one lane per row (32 rows, 32 lanes).
DEVINL void layernorm_inplace(h16* buf, int stride, int Nw,
                              const float* __restrict__ gamma,
                              const float* __restrict__ beta, int lane) {
  h16* p = buf + lane * stride;
  float s = 0.f, q = 0.f;
  for (int c = 0; c < Nw; ++c) { float v = (float)p[c]; s += v; q += v * v; }
  const float mean = s / (float)Nw;
  const float var  = q / (float)Nw - mean * mean;
  const float rstd = rsqrtf(var + 1e-5f);
  for (int c = 0; c < Nw; ++c)
    p[c] = (h16)(((float)p[c] - mean) * rstd * gamma[c] + beta[c]);
}

// Final LayerNorm: read f16 LDS, write f32 to global (Nw == 128), 1 lane/row.
DEVINL void layernorm_out(const h16* buf, int stride,
                          const float* __restrict__ gamma,
                          const float* __restrict__ beta,
                          float* __restrict__ outg, int lane) {
  const h16* p = buf + lane * stride;
  float s = 0.f, q = 0.f;
  for (int c = 0; c < 128; ++c) { float v = (float)p[c]; s += v; q += v * v; }
  const float mean = s / 128.f;
  const float var  = q / 128.f - mean * mean;
  const float rstd = rsqrtf(var + 1e-5f);
  float* o = outg + (size_t)lane * 128;
  for (int c = 0; c < 128; ++c)
    o[c] = ((float)p[c] - mean) * rstd * gamma[c] + beta[c];
}

// ---------------------------------------------------------------------------
// Weight pack kernel: f32 [slice][Ka][Na] row-major -> f16 WMMA-B tile layout
// with zero padding to [Kp][Np]. 32 threads/block; blockIdx.x = tile,
// blockIdx.y = slice (genre).
// ---------------------------------------------------------------------------
__global__ void pack_b_kernel(const float* __restrict__ W, h16* __restrict__ out,
                              int Ka, int Na, int Kp, int Np) {
  const int lane  = threadIdx.x;
  const int slice = blockIdx.y;
  const int tile  = blockIdx.x;
  const int nbt = Np >> 4;
  const int kb = tile / nbt;
  const int nb = tile - kb * nbt;
  const float* Ws = W + (size_t)slice * Ka * Na;
  h16* o = out + (size_t)slice * Kp * Np + ((size_t)tile << 9) + (lane << 4);
  const int kbase = kb * 32 + ((lane & 16) ? 8 : 0);
  const int n = nb * 16 + (lane & 15);
#pragma unroll
  for (int e = 0; e < 16; ++e) {
    const int k = kbase + e + ((e >= 8) ? 8 : 0);
    float v = (k < Ka && n < Na) ? Ws[(size_t)k * Na + n] : 0.f;
    o[e] = (h16)v;
  }
}

// ---------------------------------------------------------------------------
// Main kernel: one wave (32 threads) per block; each wave owns 32 item rows.
// ---------------------------------------------------------------------------
__global__ __launch_bounds__(32) void itemtower_kernel(
    const int* __restrict__ item_ids, const int* __restrict__ gvec,
    const int* __restrict__ years,    const float* __restrict__ title,
    const float* __restrict__ item_emb, const float* __restrict__ year_emb,
    const float* __restrict__ Wg,  const float* __restrict__ bg,
    const float* __restrict__ bt1, const float* __restrict__ bt2,
    const float* __restrict__ bb0, const float* __restrict__ g0, const float* __restrict__ be0,
    const float* __restrict__ bb1, const float* __restrict__ g1, const float* __restrict__ be1,
    const float* __restrict__ bb2, const float* __restrict__ g2, const float* __restrict__ be2,
    const float* __restrict__ ba,
    const float* __restrict__ eb1, const float* __restrict__ eb2, const float* __restrict__ eb3,
    const float* __restrict__ bagg, const float* __restrict__ bo,
    const float* __restrict__ go,   const float* __restrict__ beo,
    const h16* __restrict__ pk, float* __restrict__ out)
{
  __shared__ WaveLds L;
  const int lane = threadIdx.x;
  const int row0 = blockIdx.x * MROWS;

  // ---- title embeddings -> buf0 (f16, 32 x 384) ----
  for (int i = lane; i < MROWS * 384; i += 32) {
    const int r = i / 384, c = i - r * 384;
    L.buf0[r * BSTRIDE + c] = (h16)title[(size_t)(row0 + r) * 384 + c];
  }

  // ---- t = relu(title @ Wt1 + bt1) -> buf1[:, :192] ----
  gemm_layer(L.buf0, BSTRIDE, 384, pk + OFF_WT1, 192, L.buf1, BSTRIDE, 0, bt1, true, lane);
  // ---- text_e = t @ Wt2 + bt2 -> buf0 cols 112..207 (x assembly begins) ----
  gemm_layer(L.buf1, BSTRIDE, 192, pk + OFF_WT2, 96, L.buf0, BSTRIDE, 112, bt2, false, lane);

  // ---- item_e -> x cols 0..63 ----
  for (int i = lane; i < MROWS * 64; i += 32) {
    const int r = i >> 6, c = i & 63;
    const int id = item_ids[row0 + r];
    L.buf0[r * BSTRIDE + c] = (h16)item_emb[(size_t)id * 64 + c];
  }
  // ---- genre_e = relu(gv @ Wg + bg) -> x cols 64..95 ----
  for (int i = lane; i < MROWS * 32; i += 32) {
    const int r = i >> 5, c = i & 31;
    float s = bg[c];
    const int* gr = gvec + (size_t)(row0 + r) * 18;
#pragma unroll
    for (int j = 0; j < 18; ++j) s += (float)gr[j] * Wg[j * 32 + c];
    L.buf0[r * BSTRIDE + 64 + c] = (h16)fmaxf(s, 0.f);
  }
  // ---- year_e -> x cols 96..111 ----
  for (int i = lane; i < MROWS * 16; i += 32) {
    const int r = i >> 4, c = i & 15;
    int y = years[row0 + r] - 1919;
    y = y < 0 ? 0 : (y > 81 ? 81 : y);
    L.buf0[r * BSTRIDE + 96 + c] = (h16)year_emb[y * 16 + c];
  }
  // ---- zero K-pad cols 208..223 ----
  for (int i = lane; i < MROWS * 16; i += 32) {
    const int r = i >> 4, c = i & 15;
    L.buf0[r * BSTRIDE + 208 + c] = (h16)0.f;
  }

  // ---- base encoder: 208(224) -> 384 -> 256 -> 128, Linear+ReLU+LN ----
  gemm_layer(L.buf0, BSTRIDE, 224, pk + OFF_WB0, 384, L.buf1, BSTRIDE, 0, bb0, true, lane);
  layernorm_inplace(L.buf1, BSTRIDE, 384, g0, be0, lane);
  gemm_layer(L.buf1, BSTRIDE, 384, pk + OFF_WB1, 256, L.buf0, BSTRIDE, 0, bb1, true, lane);
  layernorm_inplace(L.buf0, BSTRIDE, 256, g1, be1, lane);
  gemm_layer(L.buf0, BSTRIDE, 256, pk + OFF_WB2, 128, L.buf1, BSTRIDE, 0, bb2, true, lane);
  layernorm_inplace(L.buf1, BSTRIDE, 128, g2, be2, lane);   // base in buf1[:, :128]

  // ---- attention logits (raw, no bias) -> float[32][32] in h1 region ----
  float* logits = (float*)L.h1;
  {
    const int cl = lane & 15;
    const int rb = (lane & 16) ? 8 : 0;
    for (int nb = 0; nb < 2; ++nb) {
      v8f acc0 = vzero8();
      v8f acc1 = vzero8();
#pragma unroll
      for (int kb = 0; kb < 4; ++kb) {
        v16h a0 = load_a(L.buf1, BSTRIDE, kb * 32, lane);
        v16h a1 = load_a(L.buf1 + 16 * BSTRIDE, BSTRIDE, kb * 32, lane);
        v16h b  = load_b(pk + OFF_WA, 2, kb, nb, lane);
        acc0 = wmma_f16(a0, b, acc0);
        acc1 = wmma_f16(a1, b, acc1);
      }
#pragma unroll
      for (int r = 0; r < 8; ++r) {
        logits[(rb + r) * 32 + nb * 16 + cl]      = acc0[r];
        logits[(16 + rb + r) * 32 + nb * 16 + cl] = acc1[r];
      }
    }
  }
  // ---- softmax over 18 genres, times gv mask -> wls[32][18] (1 lane/row) ----
  {
    const int r = lane;
    float lg[18], m = -1e30f;
#pragma unroll
    for (int g = 0; g < 18; ++g) { lg[g] = logits[r * 32 + g] + ba[g]; m = fmaxf(m, lg[g]); }
    float s = 0.f;
#pragma unroll
    for (int g = 0; g < 18; ++g) { lg[g] = __expf(lg[g] - m); s += lg[g]; }
    const float inv = 1.f / s;
    const int* gr = gvec + (size_t)(row0 + r) * 18;
#pragma unroll
    for (int g = 0; g < 18; ++g) L.wls[r * 18 + g] = lg[g] * inv * (float)gr[g];
  }

  // ---- 18 genre experts, refinement accumulated in D-fragment registers ----
  v8f ref0a = vzero8(), ref1a = vzero8();   // rows 0..15, cols 0..15 / 16..31
  v8f ref0b = vzero8(), ref1b = vzero8();   // rows 16..31
  {
    const int cl = lane & 15;
    const int rb = (lane & 16) ? 8 : 0;
    for (int g = 0; g < 18; ++g) {
      gemm_layer(L.buf1, BSTRIDE, 128, pk + OFF_E1 + (size_t)g * (128 * 64), 64,
                 L.h1, H1STRIDE, 0, eb1 + g * 64, true, lane);
      gemm_layer(L.h1, H1STRIDE, 64, pk + OFF_E2 + (size_t)g * (64 * 32), 32,
                 L.h2, H2STRIDE, 0, eb2 + g * 32, true, lane);
      float w8a[8], w8b[8];
#pragma unroll
      for (int r = 0; r < 8; ++r) {
        w8a[r] = L.wls[(rb + r) * 18 + g];
        w8b[r] = L.wls[(16 + rb + r) * 18 + g];
      }
      v16h a0 = load_a(L.h2, H2STRIDE, 0, lane);
      v16h a1 = load_a(L.h2 + 16 * H2STRIDE, H2STRIDE, 0, lane);
#pragma unroll
      for (int nb = 0; nb < 2; ++nb) {
        v16h b = load_b(pk + OFF_E3 + (size_t)g * (32 * 32), 2, 0, nb, lane);
        v8f acc0 = wmma_f16(a0, b, vzero8());
        v8f acc1 = wmma_f16(a1, b, vzero8());
        const float bv = eb3[g * 32 + nb * 16 + cl];
        v8f& ra = nb ? ref1a : ref0a;
        v8f& rbv8 = nb ? ref1b : ref0b;
#pragma unroll
        for (int r = 0; r < 8; ++r) {
          ra[r]   += (acc0[r] + bv) * w8a[r];
          rbv8[r] += (acc1[r] + bv) * w8b[r];
        }
      }
    }
    // refinements -> buf1 cols 128..159 (combined = [base | ref], K = 160)
#pragma unroll
    for (int r = 0; r < 8; ++r) {
      L.buf1[(rb + r) * BSTRIDE + 128 + cl]      = (h16)ref0a[r];
      L.buf1[(rb + r) * BSTRIDE + 144 + cl]      = (h16)ref1a[r];
      L.buf1[(16 + rb + r) * BSTRIDE + 128 + cl] = (h16)ref0b[r];
      L.buf1[(16 + rb + r) * BSTRIDE + 144 + cl] = (h16)ref1b[r];
    }
  }

  // ---- refined = relu(combined @ Wagg + bagg) -> buf0[:, :128] ----
  gemm_layer(L.buf1, BSTRIDE, 160, pk + OFF_WAGG, 128, L.buf0, BSTRIDE, 0, bagg, true, lane);
  // ---- out = LN(relu(refined @ Wo + bo)) ----
  gemm_layer(L.buf0, BSTRIDE, 128, pk + OFF_WO, 128, L.buf1, BSTRIDE, 0, bo, true, lane);
  layernorm_out(L.buf1, BSTRIDE, go, beo, out + (size_t)row0 * 128, lane);
}

// ---------------------------------------------------------------------------
// Host launch
// ---------------------------------------------------------------------------
extern "C" void kernel_launch(void* const* d_in, const int* in_sizes, int n_in,
                              void* d_out, int out_size, void* d_ws, size_t ws_size,
                              hipStream_t stream) {
  (void)n_in; (void)out_size; (void)ws_size;

  const int*   item_ids = (const int*)d_in[0];
  const int*   gvec     = (const int*)d_in[1];
  const int*   years    = (const int*)d_in[2];
  const float* title    = (const float*)d_in[3];
  const float* item_emb = (const float*)d_in[4];
  const float* year_emb = (const float*)d_in[5];
  const float* Wg   = (const float*)d_in[6];
  const float* bg   = (const float*)d_in[7];
  const float* Wt1  = (const float*)d_in[8];
  const float* bt1  = (const float*)d_in[9];
  const float* Wt2  = (const float*)d_in[10];
  const float* bt2  = (const float*)d_in[11];
  const float* Wb0  = (const float*)d_in[12];
  const float* bb0  = (const float*)d_in[13];
  const float* g0   = (const float*)d_in[14];
  const float* be0  = (const float*)d_in[15];
  const float* Wb1  = (const float*)d_in[16];
  const float* bb1  = (const float*)d_in[17];
  const float* g1   = (const float*)d_in[18];
  const float* be1  = (const float*)d_in[19];
  const float* Wb2  = (const float*)d_in[20];
  const float* bb2  = (const float*)d_in[21];
  const float* g2   = (const float*)d_in[22];
  const float* be2  = (const float*)d_in[23];
  const float* Wa   = (const float*)d_in[24];
  const float* ba   = (const float*)d_in[25];
  const float* eW1  = (const float*)d_in[26];
  const float* eb1  = (const float*)d_in[27];
  const float* eW2  = (const float*)d_in[28];
  const float* eb2  = (const float*)d_in[29];
  const float* eW3  = (const float*)d_in[30];
  const float* eb3  = (const float*)d_in[31];
  const float* Wagg = (const float*)d_in[32];
  const float* bagg = (const float*)d_in[33];
  const float* Wo   = (const float*)d_in[34];
  const float* bo   = (const float*)d_in[35];
  const float* go   = (const float*)d_in[36];
  const float* beo  = (const float*)d_in[37];

  h16* pk = (h16*)d_ws;

  auto pack = [&](const float* W, size_t off, int Ka, int Na, int Kp, int Np, int slices) {
    dim3 grid((Kp / 32) * (Np / 16), slices);
    pack_b_kernel<<<grid, 32, 0, stream>>>(W, pk + off, Ka, Na, Kp, Np);
  };

  pack(Wt1,  OFF_WT1, 384, 192, 384, 192, 1);
  pack(Wt2,  OFF_WT2, 192,  96, 192,  96, 1);
  pack(Wb0,  OFF_WB0, 208, 384, 224, 384, 1);
  pack(Wb1,  OFF_WB1, 384, 256, 384, 256, 1);
  pack(Wb2,  OFF_WB2, 256, 128, 256, 128, 1);
  pack(Wa,   OFF_WA,  128,  18, 128,  32, 1);
  pack(eW1,  OFF_E1,  128,  64, 128,  64, 18);
  pack(eW2,  OFF_E2,   64,  32,  64,  32, 18);
  pack(eW3,  OFF_E3,   32,  32,  32,  32, 18);
  pack(Wagg, OFF_WAGG,160, 128, 160, 128, 1);
  pack(Wo,   OFF_WO,  128, 128, 128, 128, 1);

  const int B = in_sizes[0];             // 131072
  const int blocks = B / MROWS;          // 32 rows per 1-wave block -> 4096
  itemtower_kernel<<<blocks, 32, 0, stream>>>(
      item_ids, gvec, years, title, item_emb, year_emb,
      Wg, bg, bt1, bt2,
      bb0, g0, be0, bb1, g1, be1, bb2, g2, be2,
      ba, eb1, eb2, eb3, bagg, bo, go, beo,
      (const h16*)pk, (float*)d_out);
}
